// GNN_19971597927185
// MI455X (gfx1250) — matrix operand order repaired and checked by reading, not compiled, not run on hardware
//
#include <hip/hip_runtime.h>

// ---------------------------------------------------------------------------
// GNN power-allocation network for MI455X (gfx1250, wave32, WMMA).
// Strategy:
//   * bf16 activations, pixel-major [P][C] with C padded to multiples of 32.
//   * All conv1x1 layers -> v_wmma_f32_16x16x32_bf16 GEMMs (M=pixels,
//     K=in-channels, N=out-channels). Weights converted once to bf16, padded.
//   * Batch chunked so the layer ping-pong buffers stay resident in the
//     192 MB L2 (HBM only touched for the 2 MB input / 2 MB output).
//   * post_users / antenna aggregation = exclude-self means, done as small
//     LDS-reduction kernels fused with the channel-concat for the next GEMM.
// ---------------------------------------------------------------------------

#define UDIM 64
#define ADIM 64
#define BDIM 128

typedef __attribute__((ext_vector_type(16))) __bf16 v16bf;
typedef __attribute__((ext_vector_type(8)))  float  v8f;

struct Frag32B { uint4 lo, hi; };  // 32 bytes == 16 bf16

static __device__ __forceinline__ v16bf load_frag(const void* a, const void* b) {
  Frag32B f;
  f.lo = *(const uint4*)a;
  f.hi = *(const uint4*)b;
  return __builtin_bit_cast(v16bf, f);
}
static __device__ __forceinline__ float  bf2f(__bf16 x) { return (float)x; }
static __device__ __forceinline__ __bf16 f2bf(float x)  { return (__bf16)x; }

// ---------------------------------------------------------------------------
// WMMA GEMM: Y[p][o] = relu?( sum_c X[p][c] * W[o][c] + bias[o] )
//   X: [Pc][K] bf16 (K in {32,64,96,128}), W: [64][K] bf16, Y: [Pc][64] bf16.
// Block = 256 threads = 8 waves; wave computes a 32-pixel x 64-channel tile
// (2 A-fragments x 4 B-fragments -> 8 accumulators, 8 WMMAs per K-step).
// Grid = Pc/256 (exact), so EXEC is all ones around every WMMA.
// ---------------------------------------------------------------------------
__global__ __launch_bounds__(256) void gemm_wmma_k(
    const __bf16* __restrict__ X, const __bf16* __restrict__ W,
    const float* __restrict__ bias, __bf16* __restrict__ Y,
    int K, int relu)
{
  const int wave = threadIdx.x >> 5;
  const int lane = threadIdx.x & 31;
  const int n    = lane & 15;   // A: row-within-tile (pixel); B: out-channel col
  const int h    = lane >> 4;   // selects K-halves per ISA 16-bit layouts
  const int p0   = blockIdx.x * 256 + wave * 32;

  v8f acc[2][4];
#pragma unroll
  for (int t = 0; t < 2; ++t)
#pragma unroll
    for (int g = 0; g < 4; ++g) {
      v8f z = {0.f, 0.f, 0.f, 0.f, 0.f, 0.f, 0.f, 0.f};
      acc[t][g] = z;
    }

  const __bf16* xrow0 = X + (size_t)(p0 + n) * K;
  const __bf16* xrow1 = xrow0 + (size_t)16 * K;

  for (int k0 = 0; k0 < K; k0 += 32) {
    // A frag (16x32 bf16): lane n holds pixel row p0+n; h=0 -> K {0..7,16..23},
    // h=1 -> K {8..15,24..31} (ISA 7.12.2 16-bit A layout).
    v16bf a0 = load_frag(xrow0 + k0 + h * 8, xrow0 + k0 + 16 + h * 8);
    v16bf a1 = load_frag(xrow1 + k0 + h * 8, xrow1 + k0 + 16 + h * 8);
    // B frag (32x16): lane n+16h holds W[g*16+n][k0+8h .. +7] and [k0+16+8h .. +7].
    const __bf16* wb = W + (size_t)n * K + k0 + h * 8;
#pragma unroll
    for (int g = 0; g < 4; ++g) {
      const __bf16* wr = wb + (size_t)g * 16 * K;
      v16bf bf = load_frag(wr, wr + 16);
      acc[0][g] = __builtin_amdgcn_wmma_f32_16x16x32_bf16(
          false, a0, false, bf, (short)0, acc[0][g], false, false);
      acc[1][g] = __builtin_amdgcn_wmma_f32_16x16x32_bf16(
          false, a1, false, bf, (short)0, acc[1][g], false, false);
    }
  }

  // Epilogue: bias + optional ReLU, store bf16 pixel-major.
  // C/D layout: lane<16 -> (M=r, N=lane); lane>=16 -> (M=r+8, N=lane-16).
#pragma unroll
  for (int t = 0; t < 2; ++t) {
    const int prow = p0 + t * 16 + h * 8;
#pragma unroll
    for (int g = 0; g < 4; ++g) {
      const float bv = bias[g * 16 + n];
#pragma unroll
      for (int r = 0; r < 8; ++r) {
        float v = acc[t][g][r] + bv;
        if (relu) v = fmaxf(v, 0.0f);
        Y[(size_t)(prow + r) * 64 + (g * 16 + n)] = f2bf(v);
      }
    }
  }
}

// ---------------------------------------------------------------------------
// post_users, in place on a [Pc][64] tensor: channels 32..63 replaced by
// (sum over other users - kept exact via (colsum - self)/63). One block per
// (chunk-batch, antenna); reduction over the 64 users in LDS.
// ---------------------------------------------------------------------------
__global__ __launch_bounds__(256) void post_users_k(__bf16* __restrict__ X)
{
  const int b = blockIdx.x >> 6;     // chunk-local batch
  const int a = blockIdx.x & 63;
  const int t = threadIdx.x;
  const int c = t & 31;              // channel offset within 32..63
  const int ug = t >> 5;             // 8 user groups

  __shared__ float part[8][32];
  __shared__ float S[32];

  float s = 0.f;
  for (int u = ug; u < UDIM; u += 8)
    s += bf2f(X[((size_t)(b * UDIM + u) * ADIM + a) * 64 + 32 + c]);
  part[ug][c] = s;
  __syncthreads();
  if (t < 32) {
    float tot = 0.f;
#pragma unroll
    for (int i = 0; i < 8; ++i) tot += part[i][c];
    S[c] = tot;
  }
  __syncthreads();
  for (int u = ug; u < UDIM; u += 8) {
    const size_t idx = ((size_t)(b * UDIM + u) * ADIM + a) * 64 + 32 + c;
    const float x = bf2f(X[idx]);
    X[idx] = f2bf((S[c] - x) * (1.0f / (UDIM - 1)));
  }
}

// ---------------------------------------------------------------------------
// Antenna aggregation fused with concat: builds gamma input [Pc][Cg]:
//   out[p][0..F-1]   = feats  (F==1: cg (f32->bf16); F==64: bf16 tensor r)
//   out[p][F..F+63]  = (sum_{v!=a} msgs[b][u][v][c]) / 63
//   out[p][F+64..Cg) = 0 padding (Cg=96 case)
// One block per (chunk-batch, user); reduce over 64 antennas in LDS.
// ---------------------------------------------------------------------------
__global__ __launch_bounds__(256) void agg_concat_k(
    const __bf16* __restrict__ msgs, const __bf16* __restrict__ featsR,
    const float* __restrict__ cg, __bf16* __restrict__ out, int Cg, int F)
{
  const int bu = blockIdx.x;         // (chunk-local b)*64 + u
  const int t  = threadIdx.x;
  const int c  = t & 63;
  const int ag = t >> 6;             // 4 antenna groups

  __shared__ float part[4][64];
  __shared__ float S[64];

  const __bf16* mrow = msgs + (size_t)bu * 64 * 64;  // [v][c], v stride 64
  float s = 0.f;
  for (int v = ag; v < ADIM; v += 4)
    s += bf2f(mrow[(size_t)v * 64 + c]);
  part[ag][c] = s;
  __syncthreads();
  if (t < 64) S[t] = part[0][t] + part[1][t] + part[2][t] + part[3][t];
  __syncthreads();

  for (int a = ag; a < ADIM; a += 4) {
    const size_t pix = (size_t)bu * 64 + a;
    __bf16* orow = out + pix * Cg;
    const float x = bf2f(mrow[(size_t)a * 64 + c]);
    orow[F + c] = f2bf((S[c] - x) * (1.0f / (ADIM - 1)));
    if (F == 1) {
      if (c == 0) orow[0] = f2bf(cg[pix]);
      if (c < Cg - 65) orow[65 + c] = f2bf(0.0f);   // pad 65..Cg-1
    } else {
      orow[c] = featsR[pix * 64 + c];               // F == 64 feature copy
    }
  }
}

// phi1 input: [Pc][32], ch0=ch1=cg, rest zero. Flat elementwise (Pc*32).
__global__ __launch_bounds__(256) void prep_phi1_k(
    const float* __restrict__ cg, __bf16* __restrict__ out)
{
  const int idx = blockIdx.x * 256 + threadIdx.x;
  const int p = idx >> 5, c = idx & 31;
  out[idx] = (c < 2) ? f2bf(cg[p]) : f2bf(0.0f);
}

// phiK input: [Pc][96], ch0=cg, ch1..64=r, ch65..95=0. Flat (Pc*96).
__global__ __launch_bounds__(256) void phi_concat_k(
    const float* __restrict__ cg, const __bf16* __restrict__ r,
    __bf16* __restrict__ out)
{
  const int idx = blockIdx.x * 256 + threadIdx.x;
  const int p = idx / 96, c = idx % 96;
  if (c == 0)       out[idx] = f2bf(cg[p]);
  else if (c < 65)  out[idx] = r[(size_t)p * 64 + (c - 1)];
  else              out[idx] = f2bf(0.0f);
}

// Final conv1x1 with 1 output channel: out[p] = dot(h[p][0..63], w) + b.
__global__ __launch_bounds__(256) void final_dot_k(
    const __bf16* __restrict__ h, const float* __restrict__ w,
    const float* __restrict__ b, float* __restrict__ out)
{
  const int p = blockIdx.x * 256 + threadIdx.x;
  float acc = b[0];
  const __bf16* hr = h + (size_t)p * 64;
#pragma unroll
  for (int c = 0; c < 64; ++c) acc += bf2f(hr[c]) * w[c];
  out[p] = acc;
}

// Weight convert f32 -> bf16 with zero K-padding: dst[O][Kp], src[O][Ks].
__global__ __launch_bounds__(256) void wprep_k(
    const float* __restrict__ src, __bf16* __restrict__ dst,
    int O, int Ks, int Kp)
{
  const int idx = blockIdx.x * 256 + threadIdx.x;
  if (idx >= O * Kp) return;
  const int o = idx / Kp, k = idx % Kp;
  dst[idx] = (k < Ks) ? f2bf(src[o * Ks + k]) : f2bf(0.0f);
}

// ---------------------------------------------------------------------------
extern "C" void kernel_launch(void* const* d_in, const int* in_sizes, int n_in,
                              void* d_out, int out_size, void* d_ws, size_t ws_size,
                              hipStream_t stream)
{
  (void)in_sizes; (void)n_in; (void)out_size;

  const float* cg      = (const float*)d_in[0];
  const float* phi1_W1 = (const float*)d_in[1];
  const float* phi1_b1 = (const float*)d_in[2];
  const float* phi1_W2 = (const float*)d_in[3];
  const float* phi1_b2 = (const float*)d_in[4];
  const float* phiK_W1 = (const float*)d_in[5];
  const float* phiK_b1 = (const float*)d_in[6];
  const float* phiK_W2 = (const float*)d_in[7];
  const float* phiK_b2 = (const float*)d_in[8];
  const float* g1_W1   = (const float*)d_in[9];
  const float* g1_b1   = (const float*)d_in[10];
  const float* g1_W2   = (const float*)d_in[11];
  const float* g1_b2   = (const float*)d_in[12];
  const float* gK_W1   = (const float*)d_in[13];
  const float* gK_b1   = (const float*)d_in[14];
  const float* gK_W2   = (const float*)d_in[15];
  const float* gK_b2   = (const float*)d_in[16];
  const float* g5_W1   = (const float*)d_in[17];
  const float* g5_b1   = (const float*)d_in[18];
  const float* g5_W2   = (const float*)d_in[19];
  const float* g5_b2   = (const float*)d_in[20];

  // ---- choose batch chunk so the activation ping-pong set fits in ws (and
  //      comfortably inside the 192 MB L2 for CB<=32).
  const size_t WEIGHT_BYTES = 204800;
  const size_t perBatch = (size_t)4096 * 2 * (96 + 64 + 64 + 128 + 64); // bytes
  int CBc = 32;
  while (CBc > 1 && WEIGHT_BYTES + 8192 + (size_t)CBc * perBatch > ws_size)
    CBc >>= 1;
  const int Pc = CBc * UDIM * ADIM;
  const int NCHUNK = BDIM / CBc;

  // ---- carve workspace
  size_t off = 0;
  auto carve = [&](size_t bytes) -> char* {
    char* p = (char*)d_ws + off;
    off += (bytes + 255) & ~(size_t)255;
    return p;
  };
  __bf16* W_phi1_1 = (__bf16*)carve(64 * 32 * 2);
  __bf16* W_phi1_2 = (__bf16*)carve(64 * 64 * 2);
  __bf16* W_phiK_1[4]; __bf16* W_phiK_2[4];
  for (int i = 0; i < 4; ++i) {
    W_phiK_1[i] = (__bf16*)carve(64 * 96 * 2);
    W_phiK_2[i] = (__bf16*)carve(64 * 64 * 2);
  }
  __bf16* W_g1_1 = (__bf16*)carve(64 * 96 * 2);
  __bf16* W_g1_2 = (__bf16*)carve(64 * 64 * 2);
  __bf16* W_gK_1[3]; __bf16* W_gK_2[3];
  for (int i = 0; i < 3; ++i) {
    W_gK_1[i] = (__bf16*)carve(64 * 128 * 2);
    W_gK_2[i] = (__bf16*)carve(64 * 64 * 2);
  }
  __bf16* W_g5_1 = (__bf16*)carve(64 * 128 * 2);

  __bf16* wA = (__bf16*)carve((size_t)Pc * 96 * 2);   // phi inputs (stride 32 or 96)
  __bf16* wH = (__bf16*)carve((size_t)Pc * 64 * 2);   // hidden
  __bf16* wM = (__bf16*)carve((size_t)Pc * 64 * 2);   // msgs
  __bf16* wG = (__bf16*)carve((size_t)Pc * 128 * 2);  // gamma inputs (96/128)
  __bf16* wR = (__bf16*)carve((size_t)Pc * 64 * 2);   // node state r

  // ---- weight conversion (tiny)
  auto wprep = [&](const float* src, __bf16* dst, int O, int Ks, int Kp) {
    const int nel = O * Kp;
    wprep_k<<<(nel + 255) / 256, 256, 0, stream>>>(src, dst, O, Ks, Kp);
  };
  wprep(phi1_W1, W_phi1_1, 64, 2, 32);
  wprep(phi1_W2, W_phi1_2, 64, 64, 64);
  for (int i = 0; i < 4; ++i) {
    wprep(phiK_W1 + (size_t)i * 64 * 65, W_phiK_1[i], 64, 65, 96);
    wprep(phiK_W2 + (size_t)i * 64 * 64, W_phiK_2[i], 64, 64, 64);
  }
  wprep(g1_W1, W_g1_1, 64, 65, 96);
  wprep(g1_W2, W_g1_2, 64, 64, 64);
  for (int i = 0; i < 3; ++i) {
    wprep(gK_W1 + (size_t)i * 64 * 128, W_gK_1[i], 64, 128, 128);
    wprep(gK_W2 + (size_t)i * 64 * 64,  W_gK_2[i], 64, 64, 64);
  }
  wprep(g5_W1, W_g5_1, 64, 128, 128);

  const dim3 blk(256);
  const int gGemm = Pc / 256;        // exact
  const int gPU   = CBc * ADIM;      // post_users blocks
  const int gAGG  = CBc * UDIM;      // agg_concat blocks

  for (int cbk = 0; cbk < NCHUNK; ++cbk) {
    const float* cgc  = cg + (size_t)cbk * Pc;
    float*       outc = (float*)d_out + (size_t)cbk * Pc;

    // ---- phi1 block: msgs = post(conv2(post(relu(conv1([cg,cg])))))
    prep_phi1_k<<<Pc * 32 / 256, blk, 0, stream>>>(cgc, wA);
    gemm_wmma_k<<<gGemm, blk, 0, stream>>>(wA, W_phi1_1, phi1_b1, wH, 32, 1);
    post_users_k<<<gPU, blk, 0, stream>>>(wH);
    gemm_wmma_k<<<gGemm, blk, 0, stream>>>(wH, W_phi1_2, phi1_b2, wM, 64, 0);
    post_users_k<<<gPU, blk, 0, stream>>>(wM);

    // ---- gamma1: r = gamma_block([cg, agg(msgs)])
    agg_concat_k<<<gAGG, blk, 0, stream>>>(wM, (const __bf16*)nullptr, cgc, wG, 96, 1);
    gemm_wmma_k<<<gGemm, blk, 0, stream>>>(wG, W_g1_1, g1_b1, wH, 96, 1);
    post_users_k<<<gPU, blk, 0, stream>>>(wH);
    gemm_wmma_k<<<gGemm, blk, 0, stream>>>(wH, W_g1_2, g1_b2, wR, 64, 1);
    post_users_k<<<gPU, blk, 0, stream>>>(wR);

    for (int i = 0; i < 4; ++i) {
      // ---- phiK block: msgs = phi_block([cg, r])
      phi_concat_k<<<Pc * 96 / 256, blk, 0, stream>>>(cgc, wR, wA);
      gemm_wmma_k<<<gGemm, blk, 0, stream>>>(wA, W_phiK_1[i], phiK_b1 + i * 64, wH, 96, 1);
      post_users_k<<<gPU, blk, 0, stream>>>(wH);
      gemm_wmma_k<<<gGemm, blk, 0, stream>>>(wH, W_phiK_2[i], phiK_b2 + i * 64, wM, 64, 0);
      post_users_k<<<gPU, blk, 0, stream>>>(wM);

      // ---- node: gamma input = [r, agg(msgs)] (128 channels)
      agg_concat_k<<<gAGG, blk, 0, stream>>>(wM, wR, (const float*)nullptr, wG, 128, 64);
      if (i < 3) {
        gemm_wmma_k<<<gGemm, blk, 0, stream>>>(wG, W_gK_1[i], gK_b1 + i * 64, wH, 128, 1);
        post_users_k<<<gPU, blk, 0, stream>>>(wH);
        gemm_wmma_k<<<gGemm, blk, 0, stream>>>(wH, W_gK_2[i], gK_b2 + i * 64, wR, 64, 1);
        post_users_k<<<gPU, blk, 0, stream>>>(wR);
      } else {
        gemm_wmma_k<<<gGemm, blk, 0, stream>>>(wG, W_g5_1, g5_b1, wH, 128, 1);
        post_users_k<<<gPU, blk, 0, stream>>>(wH);
        final_dot_k<<<Pc / 256, blk, 0, stream>>>(wH, g5_W2, g5_b2, outc);
      }
    }
  }
}
